// MultiHeadAttention_9053791060301
// MI455X (gfx1250) — matrix operand order repaired and checked by reading, not compiled, not run on hardware
//
#include <hip/hip_runtime.h>
#include <hip/hip_bf16.h>

// ---------------------------------------------------------------------------
// MHA forward for MI455X (gfx1250, wave32, WMMA 16x16x32 f16 -> f32 acc)
// B=2, S=2048, D=1024, H=16, DK=64
// CDNA5 paths: v_wmma, TDM tensor_load_to_lds (K tiles), async global->LDS
// copies (f16 GEMM staging), split wait counters.
// ---------------------------------------------------------------------------

typedef _Float16 v16h __attribute__((ext_vector_type(16)));
typedef _Float16 v8h  __attribute__((ext_vector_type(8)));
typedef float    v8f  __attribute__((ext_vector_type(8)));
typedef float    v16f __attribute__((ext_vector_type(16)));
typedef unsigned int v4u __attribute__((ext_vector_type(4)));
typedef int      v4i  __attribute__((ext_vector_type(4)));
typedef int      v8i  __attribute__((ext_vector_type(8)));

constexpr int Bn  = 2;
constexpr int Sn  = 2048;
constexpr int Dn  = 1024;
constexpr int Hn  = 16;
constexpr int DKn = 64;
constexpr int BS  = Bn * Sn;   // 4096 flattened rows

static __device__ inline v16h combine8(v8h lo, v8h hi) {
  return __builtin_shufflevector(lo, hi, 0,1,2,3,4,5,6,7,8,9,10,11,12,13,14,15);
}

// 16-element loads with on-the-fly f32 -> f16 conversion when needed
static __device__ inline v16h ld16(const _Float16* p) { return *(const v16h*)p; }
static __device__ inline v16h ld16(const float* p) {
  v16f f = *(const v16f*)p;
  return __builtin_convertvector(f, v16h);
}

static __device__ inline void st_out(float* Y, size_t i, float v)    { Y[i] = v; }
static __device__ inline void st_out(_Float16* Y, size_t i, float v) { Y[i] = (_Float16)v; }

static __device__ inline v8f wmma32(v16h a, v16h b, v8f c) {
  // (neg_a, A, neg_b, B, c_mod, C, reuse_a, reuse_b)
  return __builtin_amdgcn_wmma_f32_16x16x32_f16(false, a, false, b, (short)0, c,
                                                false, false);
}

// ---- CDNA5 async global -> LDS copy (ASYNCcnt) ----------------------------
static __device__ inline void async_ld_b128(const void* g, unsigned lds_byte_addr) {
  // the VDST VGPR holds the LDS byte address; low 32 bits of a flat LDS pointer are the
  // LDS offset (ISA 10.2 aperture rules).
  asm volatile("global_load_async_to_lds_b128 %0, %1, off"
               :: "v"(lds_byte_addr), "v"(g) : "memory");
}
static __device__ inline void wait_async0() {
#if __has_builtin(__builtin_amdgcn_s_wait_asynccnt)
  __builtin_amdgcn_s_wait_asynccnt(0);
#else
  asm volatile("s_wait_asynccnt 0x0" ::: "memory");
#endif
}
static __device__ inline void wait_tensor0() {
#if __has_builtin(__builtin_amdgcn_s_wait_tensorcnt)
  __builtin_amdgcn_s_wait_tensorcnt(0);
#else
  asm volatile("s_wait_tensorcnt 0x0" ::: "memory");
#endif
}

// ---- CDNA5 Tensor Data Mover: 2-D tile load, 2-byte elements --------------
// D# layout per ISA 8.3/8.4: group0 {count=1, lds_addr, global_addr, type=2},
// group1 {data_size=1(2B), tensor_dim0/1, tile_dim0/1, tensor_dim0_stride}.
static __device__ inline void tdm_load_tile_2d_f16(
    const void* gaddr, unsigned lds_byte_addr,
    unsigned tile_d0, unsigned tile_d1,
    unsigned tensor_d0, unsigned tensor_d1, unsigned stride0)
{
  unsigned long long ga = (unsigned long long)(size_t)gaddr;
  v4u g0;
  g0.x = 1u;                                    // count=1, user mode, no gather
  g0.y = lds_byte_addr;                         // lds_addr [63:32]
  g0.z = (unsigned)(ga & 0xffffffffu);          // global_addr [95:64]
  g0.w = (unsigned)((ga >> 32) & 0x1ffffffu)    // global_addr [120:96]
       | (2u << 30);                            // type=2 ("image") [127:126]
  v8i g1;
  g1[0] = (int)(1u << 16);                      // data_size=1 -> 2 bytes
  g1[1] = (int)(tensor_d0 << 16);               // tensor_dim0[15:0] @ [63:48]
  g1[2] = (int)((tensor_d0 >> 16) | (tensor_d1 << 16));   // dim0 hi | dim1 lo
  g1[3] = (int)((tensor_d1 >> 16) | (tile_d0 << 16));     // dim1 hi | tile_dim0
  g1[4] = (int)(tile_d1 & 0xffffu);             // tile_dim1 (tile_dim2 = 0)
  g1[5] = (int)stride0;                         // tensor_dim0_stride[31:0]
  g1[6] = 0;
  g1[7] = 0;
  v4i gz = {0, 0, 0, 0};
#if __has_include(<hip/amd_detail/amd_gfx1250_TDM.h>)
  v8i gz8 = {0, 0, 0, 0, 0, 0, 0, 0};
  __builtin_amdgcn_tensor_load_to_lds(g0, g1, gz, gz, gz8, 0);
#else
  __builtin_amdgcn_tensor_load_to_lds(g0, g1, gz, gz, 0);
#endif
}

// ---------------------------------------------------------------------------
// GEMM: Y[M,N] = X[M,K] @ W[N,K]^T + bias[N]
// Block tile 128x128, 256 threads = 8 waves (2 M x 4 N), wave tile 64x32.
// f16 sources staged via async global->LDS copies; f32 sources converted
// through VGPRs.
// ---------------------------------------------------------------------------
static __device__ inline void stage_row16(const float* g, _Float16* l) {
  *(v16h*)l = ld16(g);
}
static __device__ inline void stage_row16(const _Float16* g, _Float16* l) {
  async_ld_b128(g,     (unsigned)(size_t)l);
  async_ld_b128(g + 8, (unsigned)(size_t)(l + 8));
}

template <typename TX, typename TW, typename TOUT>
__global__ __launch_bounds__(256) void wmma_gemm_xwT(
    const TX* __restrict__ X, const TW* __restrict__ W,
    const float* __restrict__ bias, TOUT* __restrict__ Y,
    int M, int N, int K)
{
  __shared__ alignas(32) _Float16 As[128][32];
  __shared__ alignas(32) _Float16 Bs[128][32];

  const int tid   = threadIdx.x;
  const int lane  = tid & 31;
  const int wave  = tid >> 5;
  const int lrow  = lane & 15;
  const int khalf = lane >> 4;
  const int wm    = wave & 1;     // 0..1  (64-row slabs)
  const int wn    = wave >> 1;    // 0..3  (32-col slabs)
  const int bm    = blockIdx.x * 128;
  const int bn    = blockIdx.y * 128;

  v8f acc[4][2];
#pragma unroll
  for (int i = 0; i < 4; ++i)
#pragma unroll
    for (int j = 0; j < 2; ++j)
      acc[i][j] = v8f{0.f, 0.f, 0.f, 0.f, 0.f, 0.f, 0.f, 0.f};

  const int ldr = tid >> 1;         // 0..127 : tile row
  const int ldc = (tid & 1) * 16;   // 0/16   : half-row of 16 elements

  for (int k0 = 0; k0 < K; k0 += 32) {
    stage_row16(X + (size_t)(bm + ldr) * K + k0 + ldc, &As[ldr][ldc]);
    stage_row16(W + (size_t)(bn + ldr) * K + k0 + ldc, &Bs[ldr][ldc]);
    wait_async0();
    __syncthreads();

    // A fragments: lane holds row (lane&15); 16 contiguous K at khalf*16.
    v16h aF[4];
#pragma unroll
    for (int i = 0; i < 4; ++i) {
      const _Float16* ar = &As[wm * 64 + i * 16 + lrow][0];
      aF[i] = combine8(*(const v8h*)(ar + khalf * 16),
                       *(const v8h*)(ar + khalf * 16 + 8));
    }
    // B fragments: lane holds col (lane&15); 16 contiguous K at khalf*16.
    v16h bF[2];
#pragma unroll
    for (int j = 0; j < 2; ++j)
      bF[j] = *(const v16h*)&Bs[wn * 32 + j * 16 + lrow][khalf * 16];

#pragma unroll
    for (int i = 0; i < 4; ++i)
#pragma unroll
      for (int j = 0; j < 2; ++j)
        acc[i][j] = wmma32(aF[i], bF[j], acc[i][j]);

    __syncthreads();
  }

  // Epilogue: C layout — lane holds col n = lane&15; VGPR r is row r (lanes
  // 0-15) or 8+r (lanes 16-31).
#pragma unroll
  for (int i = 0; i < 4; ++i) {
#pragma unroll
    for (int j = 0; j < 2; ++j) {
      const int n = bn + wn * 32 + j * 16 + lrow;
      const float bb = bias[n];
#pragma unroll
      for (int r = 0; r < 8; ++r) {
        const int m = bm + wm * 64 + i * 16 + khalf * 8 + r;
        st_out(Y, (size_t)m * N + n, acc[i][j][r] + bb);
      }
    }
  }
}

// ---------------------------------------------------------------------------
// Flash attention: one block per (b, h, 128-row q chunk); 8 waves, each owns
// a 16-row q tile. K tile staged by the Tensor Data Mover; V tile transposed
// through VGPRs into LDS.
// ---------------------------------------------------------------------------
__global__ __launch_bounds__(256) void flash_attn_kernel(
    const _Float16* __restrict__ Q,
    const _Float16* __restrict__ Kd,
    const _Float16* __restrict__ Vd,
    const int* __restrict__ mask,     // [B, S] effectively
    _Float16* __restrict__ Oa)        // [B, S, D] head-merged attention output
{
  __shared__ alignas(32) _Float16 Ks[32][64];     // K tile  [key][d]   (TDM dest)
  __shared__ alignas(32) _Float16 Vt[64][40];     // V tile transposed [d][key]
  __shared__ alignas(32) _Float16 Ps[8][16][32];  // per-wave P round-trip buffer
  __shared__ float MB[Sn];                        // precomputed mask bias

  const int tid   = threadIdx.x;
  const int lane  = tid & 31;
  const int wave  = tid >> 5;
  const int lrow  = lane & 15;
  const int khalf = lane >> 4;

  constexpr int nqc = Sn / 128;
  const int blk = blockIdx.x;
  const int qc  = blk % nqc;
  const int h   = (blk / nqc) % Hn;
  const int b   = blk / (nqc * Hn);
  const int q0  = qc * 128 + wave * 16;

  // Mask bias table (loop-invariant per key) -> LDS once.
  for (int i = tid; i < Sn; i += 256)
    MB[i] = (mask[(size_t)b * Sn + i] == 0) ? -1e9f : 0.f;

  // Q fragments held in registers across the whole key loop (2 K-steps of 32).
  v16h qF[2];
#pragma unroll
  for (int s = 0; s < 2; ++s) {
    const _Float16* qp = Q + ((size_t)b * Sn + q0 + lrow) * Dn + h * DKn + s * 32;
    qF[s] = combine8(*(const v8h*)(qp + khalf * 16),
                     *(const v8h*)(qp + khalf * 16 + 8));
  }

  v8f acc[4];
#pragma unroll
  for (int j = 0; j < 4; ++j) acc[j] = v8f{0.f,0.f,0.f,0.f,0.f,0.f,0.f,0.f};
  float m[8], l[8];
#pragma unroll
  for (int r = 0; r < 8; ++r) { m[r] = -3.0e38f; l[r] = 0.f; }

  const int krow = tid >> 3;        // 0..31 key row (V staging)
  const int koff = (tid & 7) * 8;   // 0..56 d offset
  const _Float16* kbase = Kd + (size_t)b * Sn * Dn + h * DKn;
  const _Float16* vbase = Vd + (size_t)b * Sn * Dn + h * DKn;

  for (int kb = 0; kb < Sn / 32; ++kb) {
    // ---- K tile via Tensor Data Mover: 32 keys x 64 d, row stride D ----
    if (wave == 0) {
      tdm_load_tile_2d_f16(kbase + (size_t)(kb * 32) * Dn,
                           (unsigned)(size_t)&Ks[0][0],
                           /*tile_d0=*/DKn, /*tile_d1=*/32,
                           /*tensor_d0=*/Dn, /*tensor_d1=*/Sn,
                           /*stride0=*/Dn);
      wait_tensor0();
    }
    // ---- V tile through VGPRs, stored transposed ----
    const _Float16* vp = vbase + (size_t)(kb * 32 + krow) * Dn + koff;
    v8h vv = *(const v8h*)vp;
#pragma unroll
    for (int i = 0; i < 8; ++i) Vt[koff + i][krow] = vv[i];
    if (kb + 1 < Sn / 32) __builtin_prefetch(vp + (size_t)32 * Dn, 0, 1);
    __syncthreads();

    // ---- scores S = Q K^T / sqrt(DK) + mask  (16 x 32) ----
    v8f sc[2];
#pragma unroll
    for (int c = 0; c < 2; ++c) {
      v8f s0 = v8f{0.f,0.f,0.f,0.f,0.f,0.f,0.f,0.f};
#pragma unroll
      for (int s = 0; s < 2; ++s) {
        v16h kB = *(const v16h*)&Ks[c * 16 + lrow][s * 32 + khalf * 16];
        s0 = wmma32(qF[s], kB, s0);
      }
      sc[c] = s0;
    }
#pragma unroll
    for (int c = 0; c < 2; ++c) {
      const float mb = MB[kb * 32 + c * 16 + lrow];
#pragma unroll
      for (int r = 0; r < 8; ++r) sc[c][r] = sc[c][r] * 0.125f + mb;
    }

    // ---- online softmax (rows live across the 16 lanes of each half) ----
    float mnew[8], sca[8];
#pragma unroll
    for (int r = 0; r < 8; ++r) {
      float rm = fmaxf(sc[0][r], sc[1][r]);
      rm = fmaxf(rm, __shfl_xor(rm, 1, 32));
      rm = fmaxf(rm, __shfl_xor(rm, 2, 32));
      rm = fmaxf(rm, __shfl_xor(rm, 4, 32));
      rm = fmaxf(rm, __shfl_xor(rm, 8, 32));
      mnew[r] = fmaxf(m[r], rm);
      sca[r]  = __expf(m[r] - mnew[r]);
      m[r]    = mnew[r];
    }
#pragma unroll
    for (int r = 0; r < 8; ++r) {
      float p0 = __expf(sc[0][r] - mnew[r]);
      float p1 = __expf(sc[1][r] - mnew[r]);
      sc[0][r] = p0; sc[1][r] = p1;
      float rs = p0 + p1;
      rs += __shfl_xor(rs, 1, 32);
      rs += __shfl_xor(rs, 2, 32);
      rs += __shfl_xor(rs, 4, 32);
      rs += __shfl_xor(rs, 8, 32);
      l[r] = l[r] * sca[r] + rs;
    }
#pragma unroll
    for (int j = 0; j < 4; ++j)
#pragma unroll
      for (int r = 0; r < 8; ++r) acc[j][r] *= sca[r];

    // ---- P (C layout) -> LDS -> A-fragment layout; LDS is in-order/wave ----
#pragma unroll
    for (int c = 0; c < 2; ++c)
#pragma unroll
      for (int r = 0; r < 8; ++r)
        Ps[wave][khalf * 8 + r][c * 16 + lrow] = (_Float16)sc[c][r];
    asm volatile("s_wait_dscnt 0" ::: "memory");
    v16h pF = combine8(*(const v8h*)&Ps[wave][lrow][khalf * 16],
                       *(const v8h*)&Ps[wave][lrow][khalf * 16 + 8]);

    // ---- O += P @ V  (B frag from transposed V tile: lane=d, contiguous k) ----
#pragma unroll
    for (int j = 0; j < 4; ++j) {
      const _Float16* vr = &Vt[j * 16 + lrow][khalf * 16];
      v16h vB = combine8(*(const v8h*)vr, *(const v8h*)(vr + 8));
      acc[j] = wmma32(pF, vB, acc[j]);
    }
    __syncthreads();
  }

  // ---- normalize and write head-merged f16 output [b, q, h*64 + d] ----
#pragma unroll
  for (int j = 0; j < 4; ++j) {
#pragma unroll
    for (int r = 0; r < 8; ++r) {
      const float v = acc[j][r] / l[r];
      Oa[((size_t)b * Sn + q0 + khalf * 8 + r) * Dn + h * DKn + j * 16 + lrow] =
          (_Float16)v;
    }
  }
}

// ---------------------------------------------------------------------------
// Host-side launch
// ---------------------------------------------------------------------------
extern "C" void kernel_launch(void* const* d_in, const int* in_sizes, int n_in,
                              void* d_out, int out_size, void* d_ws, size_t ws_size,
                              hipStream_t stream) {
  (void)in_sizes; (void)n_in; (void)out_size; (void)ws_size;

  const float* queries = (const float*)d_in[0];
  const float* keys    = (const float*)d_in[1];
  const float* values  = (const float*)d_in[2];
  const int*   mask    = (const int*)d_in[3];
  const float* Wq = (const float*)d_in[4];  const float* bq = (const float*)d_in[5];
  const float* Wk = (const float*)d_in[6];  const float* bk = (const float*)d_in[7];
  const float* Wv = (const float*)d_in[8];  const float* bv = (const float*)d_in[9];
  const float* Wo = (const float*)d_in[10]; const float* bo = (const float*)d_in[11];
  float* out = (float*)d_out;

  // Workspace: q/k/v/attn in f16, each BS*D halves (8 MB) -> 32 MB total.
  _Float16* qh = (_Float16*)d_ws;
  _Float16* kh = qh + (size_t)BS * Dn;
  _Float16* vh = kh + (size_t)BS * Dn;
  _Float16* ah = vh + (size_t)BS * Dn;

  dim3 g(BS / 128, Dn / 128);   // 32 x 8 blocks
  dim3 blk(256);

  // QKV projections (f32 in, f16 out)
  wmma_gemm_xwT<<<g, blk, 0, stream>>>(queries, Wq, bq, qh, BS, Dn, Dn);
  wmma_gemm_xwT<<<g, blk, 0, stream>>>(keys,    Wk, bk, kh, BS, Dn, Dn);
  wmma_gemm_xwT<<<g, blk, 0, stream>>>(values,  Wv, bv, vh, BS, Dn, Dn);

  // Flash attention: B*H*(S/128) = 512 blocks
  flash_attn_kernel<<<dim3(Bn * Hn * (Sn / 128)), blk, 0, stream>>>(
      qh, kh, vh, mask, ah);

  // Output projection (f16 in, f32 out -> d_out)
  wmma_gemm_xwT<<<g, blk, 0, stream>>>(ah, Wo, bo, out, BS, Dn, Dn);
}
